// GatedDeltaNet_87608742903897
// MI455X (gfx1250) — compile-verified, matrix operand order checked by reading, not run on hardware
//
#include <hip/hip_runtime.h>

// Gated linear attention (GatedDeltaNet-style scalar gate), chunked WMMA kernel
// for gfx1250 (MI455X). B=4, S=2048, H=16, D=128. Chunk T=32.
// Grid: B*H*2 = 128 workgroups (d_v split by 2), block = 256 threads (8 waves).
// Pipeline: next chunk's q/k/v staged into LDS with global_load_async_to_lds_b128
// (ASYNCcnt), overlapping the DMA with all WMMA work of the current chunk.

typedef __attribute__((ext_vector_type(16))) __bf16 v16bf;
typedef __attribute__((ext_vector_type(8)))  __bf16 v8bf;
typedef __attribute__((ext_vector_type(8)))  float  v8f;

#define DEVI __device__ __forceinline__

constexpr int BB   = 4;
constexpr int SEQ  = 2048;
constexpr int NH   = 16;
constexpr int DD   = 128;     // head dim
constexpr int CT   = 32;      // chunk length
constexpr int NC   = SEQ / CT;
constexpr int FS   = 2;       // d_v split factor
constexpr int DV   = DD / FS; // 64 v-columns per workgroup

// LDS row strides (elements), padded to dodge bank conflicts; rows stay 16B aligned
constexpr int QLD  = 136;  // sQ/sK: 32 x 128 bf16
constexpr int KTLD = 40;   // sKT : 128 x 32 bf16 (k~ transposed, decay-scaled)
constexpr int VTLD = 40;   // sVT : 64 x 32 bf16 (v transposed)
constexpr int ALD  = 40;   // sA  : 32 x 32 bf16 (decayed masked attention)
constexpr int SFLD = 68;   // sSf : 128 x 64 f32 master state
constexpr int SBLD = 136;  // sST : 64 x 128 bf16 state transposed [dv][dk]

DEVI v8f wmma_bf16(v16bf a, v16bf b, v8f c) {
  return __builtin_amdgcn_wmma_f32_16x16x32_bf16(false, a, false, b, (short)0, c,
                                                 false, false);
}

// Async DMA: copy 16B per lane from global (saddr base + per-lane u32 offset)
// into LDS at per-lane byte offset. Tracked by ASYNCcnt.
DEVI void async_copy_b128(unsigned int lds_off, unsigned int goff,
                          unsigned long long base) {
  asm volatile("global_load_async_to_lds_b128 %0, %1, %2"
               :: "v"(lds_off), "v"(goff), "s"(base) : "memory");
}

// Load a 16x32 A-operand (or 32x16 B-operand stored transposed [n][k]) tile
// from LDS. Per ISA layout: lanes 0-15 -> rows, K chunks {0..7,16..23};
// lanes 16-31 -> same rows, K chunks {8..15,24..31}.
DEVI v16bf load_op(const __bf16* base, int ld, int row0, int col0, int lane) {
  const int r = lane & 15;
  const int h = lane >> 4;
  const __bf16* p = base + (row0 + r) * ld + col0;
  v8bf lo = *(const v8bf*)(p + h * 8);
  v8bf hi = *(const v8bf*)(p + 16 + h * 8);
  v16bf out;
#pragma unroll
  for (int i = 0; i < 8; ++i) { out[i] = lo[i]; out[i + 8] = hi[i]; }
  return out;
}

__global__ __launch_bounds__(256, 1)
void gdn_chunked_wmma(const float* __restrict__ q, const float* __restrict__ k,
                      const float* __restrict__ v, const float* __restrict__ ga,
                      const float* __restrict__ gb,
                      const float* __restrict__ A_log,
                      const float* __restrict__ dt_bias,
                      float* __restrict__ out) {
  __shared__ __align__(16) __bf16 sQ[CT * QLD];
  __shared__ __align__(16) __bf16 sK[CT * QLD];
  __shared__ __align__(16) __bf16 sKT[DD * KTLD];
  __shared__ __align__(16) __bf16 sVT[DV * VTLD];
  __shared__ __align__(16) __bf16 sA[CT * ALD];
  __shared__ __align__(16) float  sSf[DD * SFLD];
  __shared__ __align__(16) __bf16 sST[DV * SBLD];
  // double-buffered raw f32 staging, filled by async DMA
  __shared__ __align__(16) float  stQ[2 * CT * DD];
  __shared__ __align__(16) float  stK[2 * CT * DD];
  __shared__ __align__(16) float  stV[2 * CT * DV];
  __shared__ float c_s[CT];   // cumulative log-gate
  __shared__ float ce_s[CT];  // exp(c_t)
  __shared__ float be_s[CT];  // sigmoid(b)

  const int tid  = threadIdx.x;
  const int lane = tid & 31;
  const int wave = tid >> 5;
  const int coln = lane & 15;
  const int halfm = lane >> 4;

  const int bid = blockIdx.x;
  const int bb  = bid / (NH * FS);
  const int rem = bid % (NH * FS);
  const int h   = rem / FS;
  const int dv0 = (rem % FS) * DV;

  // zero state
  for (int i = tid; i < DD * SFLD; i += 256) sSf[i] = 0.f;
  for (int i = tid; i < DV * SBLD; i += 256) sST[i] = (__bf16)0.f;

  const unsigned int offQ = (unsigned int)(uintptr_t)(&stQ[0]);
  const unsigned int offK = (unsigned int)(uintptr_t)(&stK[0]);
  const unsigned int offV = (unsigned int)(uintptr_t)(&stV[0]);
  const unsigned long long qb = (unsigned long long)(uintptr_t)q;
  const unsigned long long kb = (unsigned long long)(uintptr_t)k;
  const unsigned long long vb = (unsigned long long)(uintptr_t)v;
  const unsigned int bhrow = (unsigned int)(bb * SEQ) * NH + (unsigned int)h;

  // stage chunk `cc` (q/k/v, f32) into staging buffer `sbuf`: 10 async b128/wave
  auto stage = [&](int sbuf, int cc) {
    const unsigned int qO = offQ + (unsigned int)(sbuf * CT * DD * 4);
    const unsigned int kO = offK + (unsigned int)(sbuf * CT * DD * 4);
    const unsigned int vO = offV + (unsigned int)(sbuf * CT * DV * 4);
#pragma unroll
    for (int i = 0; i < 4; ++i) {
      const int f = i * 1024 + tid * 4;              // flat f32 index in 32x128
      const int rr = f >> 7, col = f & 127;
      const unsigned int go =
          ((bhrow + (unsigned int)(cc * CT + rr) * NH) * DD + col) * 4u;
      async_copy_b128(qO + (unsigned int)(f * 4), go, qb);
      async_copy_b128(kO + (unsigned int)(f * 4), go, kb);
    }
#pragma unroll
    for (int i = 0; i < 2; ++i) {
      const int f = i * 1024 + tid * 4;              // flat f32 index in 32x64
      const int rr = f >> 6, col = f & 63;
      const unsigned int go =
          ((bhrow + (unsigned int)(cc * CT + rr) * NH) * DD + dv0 + col) * 4u;
      async_copy_b128(vO + (unsigned int)(f * 4), go, vb);
    }
  };

  stage(0, 0);  // prologue: chunk 0 -> buffer 0

  const float expA = __expf(A_log[h]);
  const float dtb  = dt_bias[h];
  const float scale = 0.08838834764831845f;  // 1/sqrt(128)

  const int r  = tid >> 3;          // chunk row handled for q/k/v staging reads
  const int cs = (tid & 7) << 4;    // q/k col start (16 floats)
  const int cv = (tid & 7) << 3;    // v col start (8 floats, within DV slice)

  for (int c = 0; c < NC; ++c) {
    const int s0 = c * CT;
    const int buf = c & 1;

    // kick off next chunk's DMA (clamped on last iter so the wait is uniform),
    // then wait for *this* chunk's 10 copies (in-order completion).
    {
      const int cn = (c + 1 < NC) ? (c + 1) : (NC - 1);
      stage(buf ^ 1, cn);
    }
    asm volatile("s_wait_asynccnt 0xa" ::: "memory");
    __syncthreads();  // chunk-c staging visible to all waves; prev phase-6 done

    // ---- gates: redundant per-wave 32-lane prefix scan (no barrier needed) ----
    const size_t abi = ((size_t)bb * SEQ + (s0 + lane)) * NH + h;
    const float av = ga[abi];
    const float bv = gb[abi];
    const float x  = av + dtb;
    const float sp = (x > 20.f) ? x : log1pf(__expf(x));
    float csum = -expA * sp;                 // log decay, <= 0
#pragma unroll
    for (int off = 1; off < 32; off <<= 1) {
      float n = __shfl_up(csum, off, 32);
      if (lane >= off) csum += n;
    }
    const float c31  = __shfl(csum, 31, 32); // chunk total log decay
    const float cex  = __expf(csum);
    const float fk   = __expf(c31 - csum);   // k~ row scale, <= 1
    const float beta = 1.f / (1.f + __expf(-bv));
    if (wave == 0) { c_s[lane] = csum; ce_s[lane] = cex; be_s[lane] = beta; }
    const float fkr = __shfl(fk, r, 32);     // scale for this thread's k row

    // ---- normalize q,k from staging; stage bf16 operand tiles to LDS ----
    {
      const float* qp = &stQ[buf * CT * DD + r * DD + cs];
      float a0[16];
      *(float4*)(a0 + 0)  = *(const float4*)(qp + 0);
      *(float4*)(a0 + 4)  = *(const float4*)(qp + 4);
      *(float4*)(a0 + 8)  = *(const float4*)(qp + 8);
      *(float4*)(a0 + 12) = *(const float4*)(qp + 12);
      float ss = 0.f;
#pragma unroll
      for (int i = 0; i < 16; ++i) ss += a0[i] * a0[i];
#pragma unroll
      for (int m = 1; m < 8; m <<= 1) ss += __shfl_xor(ss, m, 32);
      const float inv = 1.f / fmaxf(sqrtf(ss), 1e-12f);
#pragma unroll
      for (int i = 0; i < 16; ++i) sQ[r * QLD + cs + i] = (__bf16)(a0[i] * inv);
    }
    {
      const float* kp = &stK[buf * CT * DD + r * DD + cs];
      float a0[16];
      *(float4*)(a0 + 0)  = *(const float4*)(kp + 0);
      *(float4*)(a0 + 4)  = *(const float4*)(kp + 4);
      *(float4*)(a0 + 8)  = *(const float4*)(kp + 8);
      *(float4*)(a0 + 12) = *(const float4*)(kp + 12);
      float ss = 0.f;
#pragma unroll
      for (int i = 0; i < 16; ++i) ss += a0[i] * a0[i];
#pragma unroll
      for (int m = 1; m < 8; m <<= 1) ss += __shfl_xor(ss, m, 32);
      const float inv = 1.f / fmaxf(sqrtf(ss), 1e-12f);
#pragma unroll
      for (int i = 0; i < 16; ++i) {
        const float kn = a0[i] * inv;
        sK[r * QLD + cs + i] = (__bf16)kn;
        sKT[(cs + i) * KTLD + r] = (__bf16)(kn * fkr);  // decay-scaled, transposed
      }
    }
    {
      const float* vp = &stV[buf * CT * DV + r * DV + cv];
      float a0[8];
      *(float4*)(a0 + 0) = *(const float4*)(vp + 0);
      *(float4*)(a0 + 4) = *(const float4*)(vp + 4);
#pragma unroll
      for (int i = 0; i < 8; ++i) sVT[(cv + i) * VTLD + r] = (__bf16)a0[i];
    }
    __syncthreads();

    // ---- phase 2: waves 0-3 -> A = QK^T (decay+mask); waves 4-7 -> dS rows 4-7
    v8f accS[4];
#pragma unroll
    for (int n = 0; n < 4; ++n) accS[n] = v8f{0.f,0.f,0.f,0.f,0.f,0.f,0.f,0.f};

    if (wave < 4) {
      const int tm = wave >> 1, tj = wave & 1;
      v8f accA = v8f{0.f,0.f,0.f,0.f,0.f,0.f,0.f,0.f};
#pragma unroll
      for (int kc = 0; kc < 4; ++kc) {
        v16bf aq = load_op(sQ, QLD, 16 * tm, 32 * kc, lane);
        v16bf bk = load_op(sK, QLD, 16 * tj, 32 * kc, lane);
        accA = wmma_bf16(aq, bk, accA);
      }
#pragma unroll
      for (int rr = 0; rr < 8; ++rr) {
        const int tt = 16 * tm + rr + 8 * halfm;
        const int jj = 16 * tj + coln;
        const float val = (jj <= tt) ? accA[rr] * __expf(c_s[tt] - c_s[jj]) : 0.f;
        sA[tt * ALD + jj] = (__bf16)val;
      }
    } else {
      v16bf ak = load_op(sKT, KTLD, 16 * wave, 0, lane);
#pragma unroll
      for (int n = 0; n < 4; ++n) {
        v16bf bv2 = load_op(sVT, VTLD, 16 * n, 0, lane);
        accS[n] = wmma_bf16(ak, bv2, accS[n]);
      }
    }
    __syncthreads();

    // ---- phase 4: every wave: one 16x16 output tile; waves 0-3 also dS ----
    {
      const int tm = wave >> 2, tn = wave & 3;
      v8f oIntra = v8f{0.f,0.f,0.f,0.f,0.f,0.f,0.f,0.f};
      v8f oInter = v8f{0.f,0.f,0.f,0.f,0.f,0.f,0.f,0.f};
      {
        v16bf aa = load_op(sA, ALD, 16 * tm, 0, lane);
        v16bf bb2 = load_op(sVT, VTLD, 16 * tn, 0, lane);
        oIntra = wmma_bf16(aa, bb2, oIntra);
      }
#pragma unroll
      for (int kc = 0; kc < 4; ++kc) {
        v16bf aq = load_op(sQ, QLD, 16 * tm, 32 * kc, lane);
        v16bf bs = load_op(sST, SBLD, 16 * tn, 32 * kc, lane);
        oInter = wmma_bf16(aq, bs, oInter);
      }
      if (wave < 4) {
        v16bf ak = load_op(sKT, KTLD, 16 * wave, 0, lane);
#pragma unroll
        for (int n = 0; n < 4; ++n) {
          v16bf bv2 = load_op(sVT, VTLD, 16 * n, 0, lane);
          accS[n] = wmma_bf16(ak, bv2, accS[n]);
        }
      }
      // epilogue: o = (intra + exp(c_t)*inter) * beta_t * scale
#pragma unroll
      for (int rr = 0; rr < 8; ++rr) {
        const int tt = 16 * tm + rr + 8 * halfm;
        const int dvg = dv0 + 16 * tn + coln;
        const float o = (oIntra[rr] + ce_s[tt] * oInter[rr]) * be_s[tt] * scale;
        out[((size_t)bb * SEQ + (s0 + tt)) * NH * DD + (size_t)h * DD + dvg] = o;
      }
    }
    __syncthreads();

    // ---- phase 6: state update, wave w owns dk tile-row w ----
    {
      const float gamma = __expf(c31);
#pragma unroll
      for (int n = 0; n < 4; ++n) {
#pragma unroll
        for (int rr = 0; rr < 8; ++rr) {
          const int dk  = 16 * wave + rr + 8 * halfm;
          const int dvl = 16 * n + coln;
          const int idx = dk * SFLD + dvl;
          const float ns = gamma * sSf[idx] + accS[n][rr];
          sSf[idx] = ns;
          sST[dvl * SBLD + dk] = (__bf16)ns;
        }
      }
    }
    // next-iteration top barrier closes the loop-carried hazards
  }
}

extern "C" void kernel_launch(void* const* d_in, const int* in_sizes, int n_in,
                              void* d_out, int out_size, void* d_ws, size_t ws_size,
                              hipStream_t stream) {
  (void)in_sizes; (void)n_in; (void)out_size; (void)d_ws; (void)ws_size;
  const float* q   = (const float*)d_in[0];
  const float* k   = (const float*)d_in[1];
  const float* v   = (const float*)d_in[2];
  const float* a   = (const float*)d_in[3];
  const float* b   = (const float*)d_in[4];
  const float* Al  = (const float*)d_in[5];
  const float* dtb = (const float*)d_in[6];
  float* out = (float*)d_out;

  dim3 grid(BB * NH * FS);  // 128 workgroups
  dim3 block(256);          // 8 waves (wave32)
  gdn_chunked_wmma<<<grid, block, 0, stream>>>(q, k, v, a, b, Al, dtb, out);
}
